// MonarchAttention_55078660604531
// MI455X (gfx1250) — matrix operand order repaired and checked by reading, not compiled, not run on hardware
//
#include <hip/hip_runtime.h>
#include <hip/hip_bf16.h>

// ---------------------------------------------------------------------------
// Monarch attention for MI455X (gfx1250, wave32, WMMA).
// B=4 H=16 N=8192 D=64, b=128, m=64, T=3.
// All GEMMs via v_wmma_f32_16x16x32_f16 (f16 operands, f32 accum).
// Intermediates in workspace as f16:
//   kbar [bh][l][k][d]  (m x D slice contiguous per l)
//   qbar [bh][k][l][d]  (b x D slice contiguous per k)
//   Lmat [bh][l][i][k]  (m x m slice contiguous per l, final iter only)
//   vbar [bh][l][k][d]  (m x D slice contiguous per l)
// ws requirement: 4 * 64*128*64*64 * 2 bytes = 256 MB.
// ---------------------------------------------------------------------------

typedef __attribute__((ext_vector_type(16))) _Float16 v16h;
typedef __attribute__((ext_vector_type(8)))  float    v8f;

#define BDIM 64
#define BB   128
#define MM   64
#define SCALE 0.125f   // 1/sqrt(64)

// --- WMMA fragment loaders (documented CDNA5 16-bit layouts) ----------------
// A (16x32, MxK): lane L: M=L&15; VGPR p holds K = (p<4?0:16)+((L>=16)?8:0)+2*(p&3), pair.
// Loading this pattern from a row-major [row][k] f16 matrix also yields the
// B-operand of C = A * W^T when rows index N (K pairs contiguous -> b32 loads).
__device__ __forceinline__ v16h frag_rowmajor_AT(const _Float16* base, int row0,
                                                 int k0, int ld, int lane) {
  v16h f;
  const int r = row0 + (lane & 15);
  const int khalf = (lane >> 4) * 8;
#pragma unroll
  for (int p = 0; p < 8; ++p) {
    const int kk = k0 + ((p < 4) ? 0 : 16) + khalf + ((p & 3) << 1);
    // contiguous f16 pair, 4-byte aligned -> single ds_load_b32
    union { unsigned u; _Float16 h[2]; } cv;
    cv.u = *(const unsigned*)(base + r * ld + kk);
    f[2 * p]     = cv.h[0];
    f[2 * p + 1] = cv.h[1];
  }
  return f;
}

// B operand from row-major W (C = A * W):  B[k][n] = W[k][n].
// lane L: N=L&15; VGPR p holds K = (p<4?0:16)+((L>=16)?8:0)+2*(p&3), pair.
__device__ __forceinline__ v16h frag_rowmajor_B(const _Float16* base, int k0,
                                                int n0, int ld, int lane) {
  v16h f;
  const int n = n0 + (lane & 15);
  const int kh = (lane >> 4) * 8;
#pragma unroll
  for (int p = 0; p < 8; ++p) {
    const int kk = k0 + ((p < 4) ? 0 : 16) + kh + ((p & 3) << 1);
    f[2 * p]     = base[kk * ld + n];
    f[2 * p + 1] = base[(kk + 1) * ld + n];
  }
  return f;
}

__device__ __forceinline__ v8f wmma16(v16h a, v16h b, v8f c) {
  return __builtin_amdgcn_wmma_f32_16x16x32_f16(false, a, false, b,
                                                (short)0, c, false, false);
}

// ---------------------------------------------------------------------------
// Kernel 0: Kbar init.  kbar[bh][l][k][d] = mean_{l'} K[bh,k,l',d]  (f16)
// grid = BH*m, block = 64 (one thread per d)
// ---------------------------------------------------------------------------
__global__ void monarch_init_kbar(const float* __restrict__ key,
                                  _Float16* __restrict__ kbar) {
  const int bh = blockIdx.x / MM;
  const int k  = blockIdx.x % MM;
  const int d  = threadIdx.x;
  const float* kg = key + ((size_t)(bh * MM + k)) * BB * BDIM;
  float s = 0.f;
#pragma unroll 8
  for (int e = 0; e < BB; ++e) s += kg[e * BDIM + d];
  const _Float16 v = (_Float16)(s * (1.0f / (float)BB));
  _Float16* kb = kbar + (size_t)bh * BB * MM * BDIM + (size_t)k * BDIM + d;
#pragma unroll 4
  for (int l = 0; l < BB; ++l) kb[(size_t)l * MM * BDIM] = v;
}

// ---------------------------------------------------------------------------
// Kernel 1: L-step.  per (bh,l):
//   S = scale * Q_l (64x64) @ Kbar_l^T ; L = softmax_rows(S)
//   w[k] = colsum(L); Qbar_l = (L^T @ Q_l) * 1/(w+eps)  -> qbar ws (f16)
//   (final iter: also store L to ws)
// grid = BH*b, block = 128 (4 waves)
// ---------------------------------------------------------------------------
__global__ void monarch_lstep(const float* __restrict__ query,
                              const _Float16* __restrict__ kbar,
                              _Float16* __restrict__ qbar,
                              _Float16* __restrict__ Lmat,
                              int store_L) {
  __shared__ _Float16 Qsh[MM * BDIM];   // 8 KB  rows=i
  __shared__ _Float16 Kbsh[MM * BDIM];  // 8 KB  rows=k
  __shared__ float    Ssh[MM * MM];     // 16 KB rows=i cols=k
  __shared__ _Float16 LTsh[MM * MM];    // 8 KB  rows=k cols=i
  __shared__ float    winv[MM];

  const int bh = blockIdx.x / BB;
  const int l  = blockIdx.x % BB;
  const int tid = threadIdx.x;
  const int wave = tid >> 5;
  const int lane = tid & 31;

  const float* qg = query + (size_t)bh * (MM * BB) * BDIM + (size_t)l * BDIM;
  const _Float16* kbg = kbar + ((size_t)(bh * BB + l)) * MM * BDIM;

  // stage Q_l (strided rows, f32->f16) and Kbar_l (contiguous f16)
  for (int idx = tid; idx < MM * BDIM; idx += 128) {
    const int i = idx >> 6, d = idx & 63;
    Qsh[idx]  = (_Float16)qg[(size_t)i * BB * BDIM + d];
    Kbsh[idx] = kbg[idx];
  }
  __syncthreads();

  // S = Q @ Kbar^T : 4x4 tiles of 16x16, K=64. wave -> tile-row.
  {
    const v16h a0 = frag_rowmajor_AT(Qsh, wave * 16, 0, BDIM, lane);
    const v16h a1 = frag_rowmajor_AT(Qsh, wave * 16, 32, BDIM, lane);
#pragma unroll
    for (int tc = 0; tc < 4; ++tc) {
      v8f acc = {};
      acc = wmma16(a0, frag_rowmajor_AT(Kbsh, tc * 16, 0, BDIM, lane), acc);
      acc = wmma16(a1, frag_rowmajor_AT(Kbsh, tc * 16, 32, BDIM, lane), acc);
      const int mb = (lane >> 4) * 8, n = tc * 16 + (lane & 15);
#pragma unroll
      for (int r = 0; r < 8; ++r)
        Ssh[(wave * 16 + mb + r) * MM + n] = acc[r];
    }
  }
  __syncthreads();

  // row softmax over k; build L^T (f16) and optionally store L to global
  if (tid < MM) {
    const int i = tid;
    float mx = -3.0e38f;
    for (int k = 0; k < MM; ++k) mx = fmaxf(mx, Ssh[i * MM + k]);
    float sum = 0.f;
    for (int k = 0; k < MM; ++k) {
      const float e = __expf((Ssh[i * MM + k] - mx) * SCALE);
      Ssh[i * MM + k] = e;
      sum += e;
    }
    const float inv = 1.0f / sum;
    _Float16* Lg = Lmat + ((size_t)(bh * BB + l)) * MM * MM + (size_t)i * MM;
    for (int k = 0; k < MM; ++k) {
      const float p = Ssh[i * MM + k] * inv;
      LTsh[k * MM + i] = (_Float16)p;
      if (store_L) Lg[k] = (_Float16)p;
    }
  }
  __syncthreads();

  // winv[k] = 1/(sum_i L[i,k] + eps)
  if (tid < MM) {
    float s = 0.f;
    for (int i = 0; i < MM; ++i) s += (float)LTsh[tid * MM + i];
    winv[tid] = 1.0f / (s + 1e-6f);
  }
  __syncthreads();

  // Qbar = (L^T @ Q) * winv : rows k, cols d.  A = LT (rowmajor), B = Q rowmajor.
  {
    const v16h a0 = frag_rowmajor_AT(LTsh, wave * 16, 0, MM, lane);
    const v16h a1 = frag_rowmajor_AT(LTsh, wave * 16, 32, MM, lane);
    _Float16* qb = qbar + (size_t)bh * MM * BB * BDIM + (size_t)l * BDIM;
#pragma unroll
    for (int tc = 0; tc < 4; ++tc) {
      v8f acc = {};
      acc = wmma16(a0, frag_rowmajor_B(Qsh, 0, tc * 16, BDIM, lane), acc);
      acc = wmma16(a1, frag_rowmajor_B(Qsh, 32, tc * 16, BDIM, lane), acc);
      const int mb = (lane >> 4) * 8, n = tc * 16 + (lane & 15);
#pragma unroll
      for (int r = 0; r < 8; ++r) {
        const int k = wave * 16 + mb + r;
        qb[(size_t)k * BB * BDIM + n] = (_Float16)(acc[r] * winv[k]);
      }
    }
  }
}

// ---------------------------------------------------------------------------
// Kernel 2: R-step.  per (bh,k):
//   S = scale * Qbar_k (128x64) @ K_k^T ; R = softmax_rows(S) (in-place f16)
//   non-final: kbar[.,l,k,.] = R @ K_k   final: vbar[.,l,k,.] = R @ V_k
// grid = BH*m, block = 256 (8 waves)
// ---------------------------------------------------------------------------
__global__ void monarch_rstep(const float* __restrict__ key,
                              const float* __restrict__ value,
                              const _Float16* __restrict__ qbar,
                              _Float16* __restrict__ kbar,
                              _Float16* __restrict__ vbar,
                              int final_) {
  __shared__ _Float16 Ksh[BB * BDIM];   // 16 KB rows=e
  __shared__ _Float16 Wsh[BB * BDIM];   // 16 KB Qbar, then V (final)
  __shared__ _Float16 Ssh[BB * BB];     // 32 KB scores -> R (in place)

  const int bh = blockIdx.x / MM;
  const int k  = blockIdx.x % MM;
  const int tid = threadIdx.x;
  const int wave = tid >> 5;
  const int lane = tid & 31;

  const size_t blk = (size_t)(bh * MM + k) * BB * BDIM;
  const float* kg = key + blk;
  const float* vg = value + blk;
  const _Float16* qbg = qbar + blk;

  if (final_) __builtin_prefetch(vg + tid * 32, 0, 1);  // global_prefetch_b8

  for (int idx = tid; idx < BB * BDIM; idx += 256) {
    Ksh[idx] = (_Float16)kg[idx];
    Wsh[idx] = qbg[idx];
  }
  __syncthreads();

  // S = Qbar @ K^T : 8x8 tiles, K=64. wave -> tile-row. store f16.
  {
    const v16h a0 = frag_rowmajor_AT(Wsh, wave * 16, 0, BDIM, lane);
    const v16h a1 = frag_rowmajor_AT(Wsh, wave * 16, 32, BDIM, lane);
#pragma unroll
    for (int tc = 0; tc < 8; ++tc) {
      v8f acc = {};
      acc = wmma16(a0, frag_rowmajor_AT(Ksh, tc * 16, 0, BDIM, lane), acc);
      acc = wmma16(a1, frag_rowmajor_AT(Ksh, tc * 16, 32, BDIM, lane), acc);
      const int mb = (lane >> 4) * 8, n = tc * 16 + (lane & 15);
#pragma unroll
      for (int r = 0; r < 8; ++r)
        Ssh[(wave * 16 + mb + r) * BB + n] = (_Float16)acc[r];
    }
  }
  __syncthreads();

  // overlap: final iteration re-stages V into Wsh (Qbar reads are done)
  if (final_)
    for (int idx = tid; idx < BB * BDIM; idx += 256)
      Wsh[idx] = (_Float16)vg[idx];

  // row softmax over e (one thread per row, in place f16)
  if (tid < BB) {
    const int lrow = tid;
    float mx = -3.0e38f;
    for (int e = 0; e < BB; ++e) mx = fmaxf(mx, (float)Ssh[lrow * BB + e]);
    float sum = 0.f;
    for (int e = 0; e < BB; ++e) {
      const float ex = __expf(((float)Ssh[lrow * BB + e] - mx) * SCALE);
      Ssh[lrow * BB + e] = (_Float16)ex;
      sum += ex;
    }
    const float inv = 1.0f / sum;
    for (int e = 0; e < BB; ++e)
      Ssh[lrow * BB + e] = (_Float16)((float)Ssh[lrow * BB + e] * inv);
  }
  __syncthreads();

  // R @ (K or V): 8x4 tiles, K=128.  A = R rowmajor, B = rowmajor.
  {
    const _Float16* W = final_ ? Wsh : Ksh;
    _Float16* dst = (final_ ? vbar : kbar) +
                    (size_t)bh * BB * MM * BDIM + (size_t)k * BDIM;
    v16h a[4];
#pragma unroll
    for (int j = 0; j < 4; ++j)
      a[j] = frag_rowmajor_AT(Ssh, wave * 16, 32 * j, BB, lane);
#pragma unroll
    for (int tc = 0; tc < 4; ++tc) {
      v8f acc = {};
#pragma unroll
      for (int j = 0; j < 4; ++j)
        acc = wmma16(a[j], frag_rowmajor_B(W, 32 * j, tc * 16, BDIM, lane), acc);
      const int mb = (lane >> 4) * 8, n = tc * 16 + (lane & 15);
#pragma unroll
      for (int r = 0; r < 8; ++r) {
        const int lrow = wave * 16 + mb + r;
        dst[(size_t)lrow * MM * BDIM + n] = (_Float16)acc[r];
      }
    }
  }
}

// ---------------------------------------------------------------------------
// Kernel 3: output.  per (bh,l): out_l (64x64 f32) = L_l @ Vbar_l
// grid = BH*b, block = 128 (4 waves)
// ---------------------------------------------------------------------------
__global__ void monarch_out(const _Float16* __restrict__ Lmat,
                            const _Float16* __restrict__ vbar,
                            float* __restrict__ out) {
  __shared__ _Float16 Lsh[MM * MM];     // 8 KB rows=i
  __shared__ _Float16 Vsh[MM * BDIM];   // 8 KB rows=k

  const int bh = blockIdx.x / BB;
  const int l  = blockIdx.x % BB;
  const int tid = threadIdx.x;
  const int wave = tid >> 5;
  const int lane = tid & 31;

  const _Float16* Lg = Lmat + ((size_t)(bh * BB + l)) * MM * MM;
  const _Float16* Vg = vbar + ((size_t)(bh * BB + l)) * MM * BDIM;
  for (int idx = tid; idx < MM * MM; idx += 128) {
    Lsh[idx] = Lg[idx];
    Vsh[idx] = Vg[idx];
  }
  __syncthreads();

  const v16h a0 = frag_rowmajor_AT(Lsh, wave * 16, 0, MM, lane);
  const v16h a1 = frag_rowmajor_AT(Lsh, wave * 16, 32, MM, lane);
  float* og = out + (size_t)bh * (MM * BB) * BDIM + (size_t)l * BDIM;
#pragma unroll
  for (int tc = 0; tc < 4; ++tc) {
    v8f acc = {};
    acc = wmma16(a0, frag_rowmajor_B(Vsh, 0, tc * 16, BDIM, lane), acc);
    acc = wmma16(a1, frag_rowmajor_B(Vsh, 32, tc * 16, BDIM, lane), acc);
    const int mb = (lane >> 4) * 8, n = tc * 16 + (lane & 15);
#pragma unroll
    for (int r = 0; r < 8; ++r) {
      const int i = wave * 16 + mb + r;
      og[(size_t)i * BB * BDIM + n] = acc[r];
    }
  }
}

// ---------------------------------------------------------------------------
extern "C" void kernel_launch(void* const* d_in, const int* in_sizes, int n_in,
                              void* d_out, int out_size, void* d_ws, size_t ws_size,
                              hipStream_t stream) {
  (void)in_sizes; (void)n_in; (void)out_size; (void)ws_size;
  const int BH = 4 * 16;
  const float* q = (const float*)d_in[0];
  const float* k = (const float*)d_in[1];
  const float* v = (const float*)d_in[2];
  float* out = (float*)d_out;

  const size_t SZ = (size_t)BH * BB * MM * BDIM;  // 33,554,432 f16 elems / buffer
  _Float16* ws   = (_Float16*)d_ws;
  _Float16* kbar = ws;
  _Float16* qbar = ws + SZ;
  _Float16* Lmat = ws + 2 * SZ;
  _Float16* vbar = ws + 3 * SZ;

  monarch_init_kbar<<<BH * MM, 64, 0, stream>>>(k, kbar);
  for (int t = 0; t < 3; ++t) {
    const int fin = (t == 2);
    monarch_lstep<<<BH * BB, 128, 0, stream>>>(q, kbar, qbar, Lmat, fin);
    monarch_rstep<<<BH * MM, 256, 0, stream>>>(k, v, qbar, kbar, vbar, fin);
  }
  monarch_out<<<BH * BB, 128, 0, stream>>>(Lmat, vbar, out);
}
